// SelfAttention_27161373180613
// MI455X (gfx1250) — matrix operand order repaired and checked by reading, not compiled, not run on hardware
//
#include <hip/hip_runtime.h>
#include <hip/hip_bf16.h>
#include <math.h>

// Problem constants (match reference)
constexpr int B  = 4;
constexpr int C  = 512;
constexpr int N  = 4096;
constexpr int D  = 64;     // C8
constexpr int MT = 32;     // output-column tile per workgroup (attn kernel)
constexpr int NCH = 64;    // n-chunk (softmax/stream axis)
constexpr int PT_PITCH = NCH + 8;   // 72 bf16 = 144 B (16B aligned, bank spread)
constexpr int SS_PITCH = MT + 2;    // 34 dwords: disjoint bank sets for row halves

typedef __attribute__((ext_vector_type(16))) __bf16        v16bf;
typedef __attribute__((ext_vector_type(8)))  float         v8f;
typedef __attribute__((ext_vector_type(4)))  float         v4f;
typedef __attribute__((ext_vector_type(4)))  unsigned int  v4u;

__device__ __forceinline__ v8f wmma_bf16(v16bf a, v16bf b, v8f c) {
  // D = A(16x32 bf16) * B(32x16 bf16) + C(16x16 f32)
  return __builtin_amdgcn_wmma_f32_16x16x32_bf16(
      /*neg_a=*/false, a, /*neg_b=*/false, b,
      /*c_mod=*/(short)0, c, /*reuse_a=*/false, /*reuse_b=*/false);
}

// bf16 A-operand (16x32 layout): lane needs elements [akb, akb+16)
// of a contiguous row -> two 16-byte (b128) loads.
__device__ __forceinline__ v16bf load_a_row_bf16(const __bf16* row, int akb) {
  union { v4u u[2]; v16bf v; } t;
  t.u[0] = *reinterpret_cast<const v4u*>(row + akb);
  t.u[1] = *reinterpret_cast<const v4u*>(row + akb + 8);
  return t.v;
}

// bf16 B-operand (32x16 layout) from a TRANSPOSED P row (PT[m][k]):
// lane needs k = [kbase, kbase+16) at fixed column m -> 16 contiguous bf16
// -> two ds_load_b128.
__device__ __forceinline__ v16bf load_b_ptrow(const __bf16* ptrow, int kbase) {
  union { v4u u[2]; v16bf v; } t;
  t.u[0] = *reinterpret_cast<const v4u*>(ptrow + kbase);
  t.u[1] = *reinterpret_cast<const v4u*>(ptrow + kbase + 8);
  return t.v;
}

// A layout from a contiguous fp32 row: 4x b128 loads + pack to bf16.
__device__ __forceinline__ v16bf load_a_row_f32(const float* row, int akb) {
  v4f f0 = *reinterpret_cast<const v4f*>(row + akb);
  v4f f1 = *reinterpret_cast<const v4f*>(row + akb + 4);
  v4f f2 = *reinterpret_cast<const v4f*>(row + akb + 8);
  v4f f3 = *reinterpret_cast<const v4f*>(row + akb + 12);
  v16bf a;
#pragma unroll
  for (int e = 0; e < 4; ++e) {
    a[e]      = (__bf16)f0[e];
    a[4 + e]  = (__bf16)f1[e];
    a[8 + e]  = (__bf16)f2[e];
    a[12 + e] = (__bf16)f3[e];
  }
  return a;
}

// ---------------------------------------------------------------------------
// Kernel 1: projection  out[m,n] = sum_c W[m,c] * x[b,c,n] + bias[m]
// A = W (row-major [M,512]), B = x[b] (row-major [512,N]).
// TRANSPOSE=true stores out as [N][M] (q becomes a ready A-operand).
// grid: (N/64, M/16, B); block: 128 (4 waves), wave w owns cols n0+16*w.
// ---------------------------------------------------------------------------
template <int M, bool TRANSPOSE>
__global__ __launch_bounds__(128) void proj_kernel(
    const float* __restrict__ x, const float* __restrict__ W,
    const float* __restrict__ bias, __bf16* __restrict__ out)
{
  const int b    = blockIdx.z;
  const int m0   = blockIdx.y * 16;
  const int lane = threadIdx.x & 31;
  const int wave = threadIdx.x >> 5;
  const int n0   = blockIdx.x * 64 + wave * 16;

  const float* xb = x + (size_t)b * C * N;

  const int arow = lane & 15;          // A: row within 16
  const int akb  = (lane >> 4) * 16;   // A: K sub-base (0 or 16)
  const int bcol = lane & 15;          // B: column within 16
  const int bkb  = (lane >> 4) * 16;   // B: K sub-base (0 or 16)
  const int crow = (lane >> 4) * 8;    // C/D: row base (0 or 8)
  const int ccol = lane & 15;

  const float* Arow = W + (size_t)(m0 + arow) * C;

  v8f acc = {};
  for (int k0 = 0; k0 < C; k0 += 32) {
    v16bf a = load_a_row_f32(Arow + k0, akb);
    v16bf bm;
#pragma unroll
    for (int j = 0; j < 8; ++j) {
      const int kk = k0 + bkb + 2 * j;
      bm[2*j]   = (__bf16)xb[(size_t)kk       * N + (n0 + bcol)];
      bm[2*j+1] = (__bf16)xb[(size_t)(kk + 1) * N + (n0 + bcol)];
    }
    acc = wmma_bf16(a, bm, acc);
  }

#pragma unroll
  for (int i = 0; i < 8; ++i) {
    const int m = m0 + crow + i;
    const int n = n0 + ccol;
    const float val = acc[i] + bias[m];
    if (TRANSPOSE) out[((size_t)b * N + n) * M + m] = (__bf16)val;   // [B][N][M]
    else           out[((size_t)b * M + m) * N + n] = (__bf16)val;   // [B][M][N]
  }
}

// ---------------------------------------------------------------------------
// Kernel 2: fused column-softmax attention.
//   S[n,m]   = sum_d q[n,d] k[d,m]          (WMMA, K=64)
//   attn     = softmax over n (per column m), streamed online
//   O[c,m]  += sum_n v[c,n] P[n,m]          (WMMA, K=64 per chunk)
//   out      = gamma * O / L + x
// grid: (N/MT, B); block: 512 threads = 16 waves.
//   waves 0..7 : S tiles (4 n-subtiles x 2 m-subtiles)
//   waves 0..1 : softmax, column m split across lane pairs (l, l+16)
//   all waves  : O tiles, wave w owns channels [32w, 32w+32) x all MT cols
// ---------------------------------------------------------------------------
__global__ __launch_bounds__(512) void attn_kernel(
    const float*  __restrict__ x,
    const __bf16* __restrict__ qw,   // [B][N][D]
    const __bf16* __restrict__ kw,   // [B][D][N]
    const __bf16* __restrict__ vw,   // [B][C][N]
    const float*  __restrict__ gamma_p,
    float* __restrict__ out)
{
  __shared__ float  sS[NCH][SS_PITCH];     // raw scores, ~8.5 KB
  __shared__ __bf16 sPT[MT][PT_PITCH];     // TRANSPOSED exp(S - max), 4.5 KB
  __shared__ float  sScale[MT], sL[MT], sM[MT];

  const int b    = blockIdx.y;
  const int m0   = blockIdx.x * MT;
  const int tid  = threadIdx.x;
  const int lane = tid & 31;
  const int wave = tid >> 5;

  const __bf16* qb = qw + (size_t)b * N * D;
  const __bf16* kb = kw + (size_t)b * D * N;
  const __bf16* vb = vw + (size_t)b * C * N;

  if (tid < MT) { sM[tid] = -INFINITY; sL[tid] = 0.f; }

  const int arow = lane & 15, akb = (lane >> 4) * 16;
  const int bcol = lane & 15, bkb = (lane >> 4) * 16;
  const int crow = (lane >> 4) * 8, ccol = lane & 15;

  const bool isS  = (wave < 8);
  const int  nSub = wave & 3;          // n sub-tile for S waves
  const int  mSub = (wave >> 2) & 1;   // m sub-tile for S waves
  const int  c0   = wave * 32;         // channel base for O tiles

  // Preload k B-operand tiles (invariant over the n loop) for S waves.
  v16bf kB0{}, kB1{};
  if (isS) {
    const int m = m0 + mSub * 16 + bcol;
#pragma unroll
    for (int j = 0; j < 8; ++j) {
      const int d = bkb + 2 * j;
      kB0[2*j]   = kb[(size_t)d        * N + m];
      kB0[2*j+1] = kb[(size_t)(d + 1)  * N + m];
      kB1[2*j]   = kb[(size_t)(d + 32) * N + m];
      kB1[2*j+1] = kb[(size_t)(d + 33) * N + m];
    }
  }

  v8f oacc[2][2] = {};  // [c_sub][m_sub], 32 fp32 VGPRs/lane

  for (int n0 = 0; n0 < N; n0 += NCH) {
    if (n0 + NCH < N) {  // hint next chunk toward the caches
      __builtin_prefetch(vb + (size_t)(c0 + arow) * N + (n0 + NCH), 0, 1);
      if (isS) __builtin_prefetch(qb + (size_t)(n0 + NCH + nSub*16 + arow) * D, 0, 1);
    }

    // A-operands for the O update: v rows are contiguous bf16 -> b128 loads.
    v16bf va[2][2];
#pragma unroll
    for (int cs = 0; cs < 2; ++cs) {
      const __bf16* vr = vb + (size_t)(c0 + cs*16 + arow) * N + n0;
      va[cs][0] = load_a_row_bf16(vr,      akb);
      va[cs][1] = load_a_row_bf16(vr + 32, akb);
    }

    if (isS) {
      const __bf16* qr = qb + (size_t)(n0 + nSub*16 + arow) * D;
      v16bf qa0 = load_a_row_bf16(qr,      akb);
      v16bf qa1 = load_a_row_bf16(qr + 32, akb);
      v8f s = {};
      s = wmma_bf16(qa0, kB0, s);
      s = wmma_bf16(qa1, kB1, s);
#pragma unroll
      for (int i = 0; i < 8; ++i)
        sS[nSub*16 + crow + i][mSub*16 + ccol] = s[i];
    }
    __syncthreads();

    // Online column softmax: waves 0-1; column m = 16*wave + (lane&15),
    // rows split between lane and lane^16, combined with wave32 shuffles.
    if (wave < 2) {
      const int m  = wave * 16 + (lane & 15);
      const int r0 = (lane >> 4) * 32;          // this lane's row half
      const float runM = sM[m];
      float cmax = -INFINITY;
#pragma unroll 8
      for (int i = 0; i < 32; ++i) cmax = fmaxf(cmax, sS[r0 + i][m]);
      cmax = fmaxf(cmax, __shfl_xor(cmax, 16));
      const float newM  = fmaxf(runM, cmax);
      const float scale = __expf(runM - newM);  // 0 on first chunk (-inf)
      float psum = 0.f;
      union { v4u u[4]; __bf16 e[32]; } pbuf;
#pragma unroll 8
      for (int i = 0; i < 32; ++i) {
        const float p = __expf(sS[r0 + i][m] - newM);
        psum += p;
        pbuf.e[i] = (__bf16)p;
      }
      // contiguous PT row segment -> 4x ds_store_b128
#pragma unroll
      for (int s = 0; s < 4; ++s)
        *reinterpret_cast<v4u*>(&sPT[m][r0 + 8 * s]) = pbuf.u[s];
      psum += __shfl_xor(psum, 16);
      if (lane < 16) {
        sM[m]     = newM;
        sL[m]     = sL[m] * scale + psum;
        sScale[m] = scale;
      }
    }
    __syncthreads();

    // Rescale running accumulators (one column per lane -> scalar multiply).
    float scl0 = sScale[ccol];
    float scl1 = sScale[16 + ccol];
#pragma unroll
    for (int cs = 0; cs < 2; ++cs)
#pragma unroll
      for (int i = 0; i < 8; ++i) {
        oacc[cs][0][i] *= scl0;
        oacc[cs][1][i] *= scl1;
      }

    // B-operands from transposed P rows: contiguous -> ds_load_b128 pairs.
    const __bf16* pt0 = &sPT[bcol][0];
    const __bf16* pt1 = &sPT[16 + bcol][0];
    v16bf pb[2][2];
    pb[0][0] = load_b_ptrow(pt0, bkb);
    pb[0][1] = load_b_ptrow(pt0, bkb + 32);
    pb[1][0] = load_b_ptrow(pt1, bkb);
    pb[1][1] = load_b_ptrow(pt1, bkb + 32);

#pragma unroll
    for (int cs = 0; cs < 2; ++cs)
#pragma unroll
      for (int ms = 0; ms < 2; ++ms) {
        oacc[cs][ms] = wmma_bf16(va[cs][0], pb[ms][0], oacc[cs][ms]);
        oacc[cs][ms] = wmma_bf16(va[cs][1], pb[ms][1], oacc[cs][ms]);
      }
    __syncthreads();  // protect sS/sPT before next chunk
  }

  // Epilogue: out = gamma * O / L + x
  const float g = gamma_p[0];
  const float linv0 = 1.f / sL[ccol];
  const float linv1 = 1.f / sL[16 + ccol];
  const float* xb2 = x   + (size_t)b * C * N;
  float*       ob  = out + (size_t)b * C * N;
#pragma unroll
  for (int cs = 0; cs < 2; ++cs)
#pragma unroll
    for (int ms = 0; ms < 2; ++ms) {
      const float linv = ms ? linv1 : linv0;
#pragma unroll
      for (int i = 0; i < 8; ++i) {
        const int c = c0 + cs*16 + crow + i;
        const int m = m0 + ms*16 + ccol;
        ob[(size_t)c * N + m] = g * oacc[cs][ms][i] * linv + xb2[(size_t)c * N + m];
      }
    }
}

// ---------------------------------------------------------------------------
extern "C" void kernel_launch(void* const* d_in, const int* in_sizes, int n_in,
                              void* d_out, int out_size, void* d_ws, size_t ws_size,
                              hipStream_t stream) {
  const float* x     = (const float*)d_in[0];
  const float* Wq    = (const float*)d_in[1];
  const float* bq    = (const float*)d_in[2];
  const float* Wk    = (const float*)d_in[3];
  const float* bk    = (const float*)d_in[4];
  const float* Wv    = (const float*)d_in[5];
  const float* bv    = (const float*)d_in[6];
  const float* gamma = (const float*)d_in[7];
  float* out = (float*)d_out;

  // bf16 workspace: q [B][N][D], k [B][D][N], v [B][C][N]  (~21 MB total)
  __bf16* q_ws = reinterpret_cast<__bf16*>(d_ws);
  __bf16* k_ws = q_ws + (size_t)B * N * D;
  __bf16* v_ws = k_ws + (size_t)B * D * N;

  dim3 pblk(128);
  proj_kernel<D, true ><<<dim3(N/64, D/16, B), pblk, 0, stream>>>(x, Wq, bq, q_ws);
  proj_kernel<D, false><<<dim3(N/64, D/16, B), pblk, 0, stream>>>(x, Wk, bk, k_ws);
  proj_kernel<C, false><<<dim3(N/64, C/16, B), pblk, 0, stream>>>(x, Wv, bv, v_ws);

  attn_kernel<<<dim3(N/MT, B), dim3(512), 0, stream>>>(
      x, q_ws, k_ws, v_ws, gamma, out);
}
